// SelfAttention_73641509257744
// MI455X (gfx1250) — matrix operand order repaired and checked by reading, not compiled, not run on hardware
//
#include <hip/hip_runtime.h>
#include <math.h>

typedef __attribute__((ext_vector_type(16))) _Float16 v16h;
typedef __attribute__((ext_vector_type(8)))  _Float16 v8h;
typedef __attribute__((ext_vector_type(8)))  float    v8f;
typedef __attribute__((ext_vector_type(4)))  float    v4f;
typedef __attribute__((ext_vector_type(4)))  int      v4i;

#define CC    512
#define CKK   64
#define NN    4096
#define BB    2
#define MROWS 640   // CK(f) + CK(g) + C(h) stacked projection rows
#define LDF   72    // 64 K + 8 pad (halves) -> 144B rows, 16B aligned
#define TILE  (64 * LDF)

// ---------------- WMMA fragment helpers (wave32, 16x16x32 f16) ----------------
// ISA 7.12.2: A 16x32 f16 lane={half,M}: e -> K = 16*(e>>3) + 8*half + (e&7)
//             B 32x16 f16 lane={half,N}: same K order as A; this code feeds A and B K = 16*half + e
//             C/D 16x16 f32: reg r -> M = r + 8*half, N = lane&15
// A tiles stored plain, like B (no block permutation); in each 32-K
// fragment each lane reads 16 contiguous halves, K = 16*half + e -> 2x ds_load_b128.

__device__ __forceinline__ int swzA(int g) {  // swap low two block-index bits
  return ((g & 1) << 1) | ((g >> 1) & 1) | (g & ~3);
}

__device__ __forceinline__ v16h ld_frag(const _Float16* p) {
  v8h lo = *(const v8h*)(p);
  v8h hi = *(const v8h*)(p + 8);
  return __builtin_shufflevector(lo, hi, 0, 1, 2, 3, 4, 5, 6, 7, 8, 9, 10, 11,
                                 12, 13, 14, 15);
}

__device__ __forceinline__ v8f wmma_f16(v16h a, v16h b, v8f c) {
  return __builtin_amdgcn_wmma_f32_16x16x32_f16(false, a, false, b, (short)0, c,
                                                false, false);
}

// ------------- async global->LDS 16B copy (ASYNCcnt path, gfx1250) ----------
#if __has_builtin(__builtin_amdgcn_global_load_async_to_lds_b128)
#define HAS_ASYNC_LDS 1
__device__ __forceinline__ void cp16(_Float16* dst, const _Float16* src) {
  __builtin_amdgcn_global_load_async_to_lds_b128((v4i*)src, (v4i*)dst, 0, 0);
}
#else
#define HAS_ASYNC_LDS 0
__device__ __forceinline__ void cp16(_Float16* dst, const _Float16* src) {
  *(v8h*)dst = *(const v8h*)src;
}
#endif

__device__ __forceinline__ void async_wait() {
#if HAS_ASYNC_LDS
#if __has_builtin(__builtin_amdgcn_s_wait_asynccnt)
  __builtin_amdgcn_s_wait_asynccnt(0);
#else
  asm volatile("s_wait_asynccnt 0" ::: "memory");
#endif
#endif
}

// stage a 64x64 f16 tile (rows k-fast, stride srcStride) into LDS [64][LDF]
template <bool SWZ>
__device__ __forceinline__ void stage64(_Float16* dst, const _Float16* src,
                                        size_t srcStride, int tid) {
#pragma unroll
  for (int i = tid; i < 512; i += 128) {
    int r = i >> 3, g = i & 7;
    cp16(dst + r * LDF + g * 8, src + r * srcStride + g * 8);
  }
}

// ---------------- f32 -> f16 conversion (weights) ----------------
__global__ void cvt_kernel(const float* __restrict__ src,
                           _Float16* __restrict__ dst, int n) {
  int i = blockIdx.x * blockDim.x + threadIdx.x;
  if (i < n) dst[i] = (_Float16)src[i];
}

// ---------------- x[b][c][n] f32 -> xT[b][n][c] f16 (tiled transpose) --------
__global__ __launch_bounds__(256) void xpose_kernel(const float* __restrict__ x,
                                                    _Float16* __restrict__ xT) {
  __shared__ float tile[32][33];
  const int b = blockIdx.z;
  const int n0 = blockIdx.x * 32, c0 = blockIdx.y * 32;
  const int tx = threadIdx.x & 31, ty = threadIdx.x >> 5;  // 32 x 8
  const float* xb = x + (size_t)b * CC * NN;
  _Float16* xtb = xT + (size_t)b * NN * CC;
#pragma unroll
  for (int r = 0; r < 32; r += 8)
    tile[ty + r][tx] = xb[(size_t)(c0 + ty + r) * NN + n0 + tx];
  __syncthreads();
#pragma unroll
  for (int r = 0; r < 32; r += 8)
    xtb[(size_t)(n0 + ty + r) * CC + c0 + tx] = (_Float16)tile[tx][ty + r];
}

// -------- fused projection GEMM: P = Wcat(640x512) * x(512x4096) + bias ------
// A = Wcat rows (k-fast, plain); B = xT rows (k-fast, plain). K-step 64,
// double-buffered async staging.
__global__ __launch_bounds__(128) void proj_kernel(
    const _Float16* __restrict__ wh, const _Float16* __restrict__ xT,
    const float* __restrict__ f_b, const float* __restrict__ g_b,
    const float* __restrict__ h_b, _Float16* __restrict__ fxT,
    _Float16* __restrict__ gxT, _Float16* __restrict__ hxh) {
  __shared__ _Float16 sA[2][TILE];
  __shared__ _Float16 sB[2][TILE];
  const int b = blockIdx.z, m0 = blockIdx.y * 64, n0 = blockIdx.x * 64;
  const int tid = threadIdx.x, wave = tid >> 5, lane = tid & 31;
  const int h = lane >> 4, nl = lane & 15;
  const _Float16* xtb = xT + (size_t)b * NN * CC;

  stage64<true>(sA[0], wh + (size_t)m0 * CC, CC, tid);
  stage64<false>(sB[0], xtb + (size_t)n0 * CC, CC, tid);

  v8f acc[4] = {};
  int t = 0;
  for (int k0 = 0; k0 < CC; k0 += 64, t ^= 1) {
    async_wait();
    __syncthreads();
    if (k0 + 64 < CC) {
      stage64<true>(sA[t ^ 1], wh + (size_t)m0 * CC + k0 + 64, CC, tid);
      stage64<false>(sB[t ^ 1], xtb + (size_t)n0 * CC + k0 + 64, CC, tid);
    }
    v16h a0 = ld_frag(sA[t] + (wave * 16 + nl) * LDF + 16 * h);
    v16h a1 = ld_frag(sA[t] + (wave * 16 + nl) * LDF + 32 + 16 * h);
#pragma unroll
    for (int jt = 0; jt < 4; ++jt) {
      v16h b0 = ld_frag(sB[t] + (jt * 16 + nl) * LDF + 16 * h);
      v16h b1 = ld_frag(sB[t] + (jt * 16 + nl) * LDF + 32 + 16 * h);
      acc[jt] = wmma_f16(a0, b0, acc[jt]);
      acc[jt] = wmma_f16(a1, b1, acc[jt]);
    }
  }
#pragma unroll
  for (int jt = 0; jt < 4; ++jt) {
#pragma unroll
    for (int r = 0; r < 8; ++r) {
      int row = m0 + wave * 16 + r + 8 * h;  // output channel
      int col = n0 + jt * 16 + nl;           // spatial position
      float v = acc[jt][r];
      if (row < CKK) {
        fxT[((size_t)b * NN + col) * CKK + row] = (_Float16)(v + f_b[row]);
      } else if (row < 2 * CKK) {
        gxT[((size_t)b * NN + col) * CKK + (row - CKK)] =
            (_Float16)(v + g_b[row - CKK]);
      } else {
        hxh[((size_t)b * CC + (row - 2 * CKK)) * NN + col] =
            (_Float16)(v + h_b[row - 2 * CKK]);
      }
    }
  }
}

// ---------------- energy + softmax: attn = softmax_j(fx^T gx) ----------------
__global__ __launch_bounds__(128) void attn_kernel(
    const _Float16* __restrict__ fxT, const _Float16* __restrict__ gxT,
    float* __restrict__ attn) {
  __shared__ _Float16 sF[TILE];     // A tile [i-local][K], plain, once
  __shared__ _Float16 sG[2][TILE];  // B tile [j-local][K], double-buffered
  __shared__ float sStat[4][16][2];
  const int b = blockIdx.y, i0 = blockIdx.x * 64;
  const int tid = threadIdx.x, wave = tid >> 5, lane = tid & 31;
  const int h = lane >> 4, nl = lane & 15;
  const _Float16* fb = fxT + (size_t)b * NN * CKK;
  const _Float16* gb = gxT + (size_t)b * NN * CKK;
  float* ab = attn + (size_t)b * NN * NN;

  stage64<true>(sF, fb + (size_t)i0 * CKK, CKK, tid);
  stage64<false>(sG[0], gb, CKK, tid);
  async_wait();
  __syncthreads();
  v16h aF0 = ld_frag(sF + (wave * 16 + nl) * LDF + 16 * h);
  v16h aF1 = ld_frag(sF + (wave * 16 + nl) * LDF + 32 + 16 * h);

  float runm[8], runs[8];
#pragma unroll
  for (int r = 0; r < 8; ++r) { runm[r] = -3.0e38f; runs[r] = 0.0f; }

  int t = 0;
  for (int j0 = 0; j0 < NN; j0 += 64, t ^= 1) {
    if (j0 + 64 < NN)
      stage64<false>(sG[t ^ 1], gb + (size_t)(j0 + 64) * CKK, CKK, tid);
#pragma unroll
    for (int jt = 0; jt < 4; ++jt) {
      v16h b0 = ld_frag(sG[t] + (jt * 16 + nl) * LDF + 16 * h);
      v16h b1 = ld_frag(sG[t] + (jt * 16 + nl) * LDF + 32 + 16 * h);
      v8f e = {};
      e = wmma_f16(aF0, b0, e);
      e = wmma_f16(aF1, b1, e);
#pragma unroll
      for (int r = 0; r < 8; ++r) {
        int row = i0 + wave * 16 + r + 8 * h;
        int col = j0 + jt * 16 + nl;
        float ev = e[r];
        ab[(size_t)row * NN + col] = ev;  // raw energy; rescaled in pass 2
        float nm = fmaxf(runm[r], ev);
        runs[r] = runs[r] * __expf(runm[r] - nm) + __expf(ev - nm);
        runm[r] = nm;
      }
    }
    async_wait();
    __syncthreads();
  }
  // reduce (max,sum) across the 16 lanes sharing each row (halves disjoint)
#pragma unroll
  for (int r = 0; r < 8; ++r) {
    float m = runm[r], s = runs[r];
#pragma unroll
    for (int d = 1; d < 16; d <<= 1) {
      float mo = __shfl_xor(m, d, 32);
      float so = __shfl_xor(s, d, 32);
      float nm = fmaxf(m, mo);
      s = s * __expf(m - nm) + so * __expf(mo - nm);
      m = nm;
    }
    if (nl == 0) {
      sStat[wave][r + 8 * h][0] = m;
      sStat[wave][r + 8 * h][1] = s;
    }
  }
  __syncthreads();
  // pass 2: coalesced float4 in-place normalization of this wave's 16 rows
  for (int m = 0; m < 16; ++m) {
    int row = i0 + wave * 16 + m;
    float rm = sStat[wave][m][0];
    float ri = 1.0f / sStat[wave][m][1];
    float* rp = ab + (size_t)row * NN;
    for (int c = lane * 4; c < NN; c += 128) {
      v4f t2 = *(const v4f*)(rp + c);
#pragma unroll
      for (int e = 0; e < 4; ++e) t2[e] = __expf(t2[e] - rm) * ri;
      *(v4f*)(rp + c) = t2;
    }
  }
}

// ---------------- out = gamma * (hx(512x4096) @ attn^T) + x ----------------
// A = hx rows (j-fast, plain, async); B = attn rows (j-fast) f32->f16.
// Double-buffered.
__global__ __launch_bounds__(128) void out_kernel(
    const _Float16* __restrict__ hxh, const float* __restrict__ attn,
    const float* __restrict__ x, const float* __restrict__ gamma,
    float* __restrict__ out) {
  __shared__ _Float16 sA[2][TILE];
  __shared__ _Float16 sB[2][TILE];
  const int b = blockIdx.z, c0 = blockIdx.y * 64, i0 = blockIdx.x * 64;
  const int tid = threadIdx.x, wave = tid >> 5, lane = tid & 31;
  const int h = lane >> 4, nl = lane & 15;
  const _Float16* hb = hxh + (size_t)b * CC * NN;
  const float* ab = attn + (size_t)b * NN * NN;

  // manual f32 -> f16 conversion staging for the attention tile
  auto stageB = [&](_Float16* dst, int j0) {
#pragma unroll
    for (int i = tid; i < 512; i += 128) {
      int r = i >> 3, g = i & 7;
      const float* src = &ab[(size_t)(i0 + r) * NN + j0 + g * 8];
      v4f t0 = *(const v4f*)(src);
      v4f t1 = *(const v4f*)(src + 4);
      v8h q;
#pragma unroll
      for (int e = 0; e < 4; ++e) {
        q[e] = (_Float16)t0[e];
        q[e + 4] = (_Float16)t1[e];
      }
      *(v8h*)&dst[r * LDF + g * 8] = q;
    }
  };

  stage64<true>(sA[0], hb + (size_t)c0 * NN, NN, tid);
  stageB(sB[0], 0);

  v8f acc[4] = {};
  int t = 0;
  for (int j0 = 0; j0 < NN; j0 += 64, t ^= 1) {
    async_wait();
    __syncthreads();
    if (j0 + 64 < NN) {
      stage64<true>(sA[t ^ 1], hb + (size_t)c0 * NN + j0 + 64, NN, tid);
      stageB(sB[t ^ 1], j0 + 64);
    }
    v16h a0 = ld_frag(sA[t] + (wave * 16 + nl) * LDF + 16 * h);
    v16h a1 = ld_frag(sA[t] + (wave * 16 + nl) * LDF + 32 + 16 * h);
#pragma unroll
    for (int jt = 0; jt < 4; ++jt) {
      v16h b0 = ld_frag(sB[t] + (jt * 16 + nl) * LDF + 16 * h);
      v16h b1 = ld_frag(sB[t] + (jt * 16 + nl) * LDF + 32 + 16 * h);
      acc[jt] = wmma_f16(a0, b0, acc[jt]);
      acc[jt] = wmma_f16(a1, b1, acc[jt]);
    }
  }
  const float g = gamma[0];
#pragma unroll
  for (int jt = 0; jt < 4; ++jt) {
#pragma unroll
    for (int r = 0; r < 8; ++r) {
      int c = c0 + wave * 16 + r + 8 * h;
      int i = i0 + jt * 16 + nl;
      size_t idx = ((size_t)b * CC + c) * NN + i;
      out[idx] = g * acc[jt][r] + x[idx];
    }
  }
}

// ---------------- launcher ----------------
extern "C" void kernel_launch(void* const* d_in, const int* in_sizes, int n_in,
                              void* d_out, int out_size, void* d_ws,
                              size_t ws_size, hipStream_t stream) {
  (void)in_sizes; (void)n_in; (void)out_size; (void)ws_size;
  const float* x     = (const float*)d_in[0];
  const float* f_w   = (const float*)d_in[1];
  const float* f_b   = (const float*)d_in[2];
  const float* g_w   = (const float*)d_in[3];
  const float* g_b   = (const float*)d_in[4];
  const float* h_w   = (const float*)d_in[5];
  const float* h_b   = (const float*)d_in[6];
  const float* gamma = (const float*)d_in[7];

  float* out  = (float*)d_out;
  float* attn = out + (size_t)BB * CC * NN;  // outputs: [out | attention]

  _Float16* xT  = (_Float16*)d_ws;              // [B][N][C]   (k-fast)
  _Float16* wh  = xT + (size_t)BB * NN * CC;    // [640][512]  stacked weights
  _Float16* fxT = wh + (size_t)MROWS * CC;      // [B][N][64]  (k-fast)
  _Float16* gxT = fxT + (size_t)BB * NN * CKK;  // [B][N][64]  (k-fast)
  _Float16* hxh = gxT + (size_t)BB * NN * CKK;  // [B][512][N] (j-fast)

  { int n = CKK * CC; cvt_kernel<<<(n + 255) / 256, 256, 0, stream>>>(f_w, wh, n); }
  { int n = CKK * CC; cvt_kernel<<<(n + 255) / 256, 256, 0, stream>>>(g_w, wh + (size_t)CKK * CC, n); }
  { int n = CC * CC;  cvt_kernel<<<(n + 255) / 256, 256, 0, stream>>>(h_w, wh + (size_t)2 * CKK * CC, n); }
  xpose_kernel<<<dim3(NN / 32, CC / 32, BB), 256, 0, stream>>>(x, xT);

  proj_kernel<<<dim3(NN / 64, MROWS / 64, BB), 128, 0, stream>>>(
      wh, xT, f_b, g_b, h_b, fxT, gxT, hxh);
  attn_kernel<<<dim3(NN / 64, BB), 128, 0, stream>>>(fxT, gxT, attn);
  out_kernel<<<dim3(NN / 64, CC / 64, BB), 128, 0, stream>>>(
      hxh, attn, x, gamma, out);
}
